// S2VT_24996709662936
// MI455X (gfx1250) — matrix operand order repaired and checked by reading, not compiled, not run on hardware
//
#include <hip/hip_runtime.h>
#include <hip/hip_bf16.h>
#include <cstdint>
#include <cstddef>

// ---------------------------------------------------------------------------
// S2VT seq2seq LSTM for MI455X (gfx1250, wave32, WMMA bf16).
// ---------------------------------------------------------------------------

typedef __attribute__((ext_vector_type(16))) __bf16 bf16x16;
typedef __attribute__((ext_vector_type(8)))  __bf16 bf16x8;
typedef __attribute__((ext_vector_type(8)))  float  f32x8;

#define NINP   4096
#define NHID   512
#define H4     2048
#define NTOK   50257
#define NTOK_PAD 50304   // 786 N-groups * 64 cols: every 16x64 wave tile is full
#define BSZ    32
#define T_ENC  80
#define T_DEC  40
#define T_ALL  120

static __device__ __forceinline__ __bf16 f2bf(float f) {
  unsigned u = __float_as_uint(f);
  unsigned r = (u + 0x7FFFu + ((u >> 16) & 1u)) >> 16;  // RNE
  return __builtin_bit_cast(__bf16, (unsigned short)r);
}

static __device__ __forceinline__ float sigm(float x) {
  return 1.0f / (1.0f + __expf(-x));
}

static __device__ __forceinline__ bf16x16 join16(bf16x8 lo, bf16x8 hi) {
  return __builtin_shufflevector(lo, hi, 0,1,2,3,4,5,6,7,8,9,10,11,12,13,14,15);
}

// ---------------------------------------------------------------------------
// fp32 -> bf16 conversion kernels
// ---------------------------------------------------------------------------
__global__ void cvt_bf16(const float* __restrict__ s, __bf16* __restrict__ d, long n) {
  long i = blockIdx.x * (long)blockDim.x + threadIdx.x;
  long st = (long)gridDim.x * blockDim.x;
  for (; i < n; i += st) d[i] = f2bf(s[i]);
}

__global__ void cvt_bf16_pad_rows(const float* __restrict__ s, __bf16* __restrict__ d,
                                  int rows, int cols, int padRows) {
  long n = (long)padRows * cols;
  long i = blockIdx.x * (long)blockDim.x + threadIdx.x;
  long st = (long)gridDim.x * blockDim.x;
  for (; i < n; i += st) {
    long r = i / cols;
    d[i] = (r < rows) ? f2bf(s[i]) : f2bf(0.0f);
  }
}

// Fill decoder region of Zx with bias only (x is zeros there). nmask = n-1, n pow2.
__global__ void fill_bias(float* __restrict__ Z, const float* __restrict__ b,
                          long total, int nmask) {
  long i = blockIdx.x * (long)blockDim.x + threadIdx.x;
  long st = (long)gridDim.x * blockDim.x;
  for (; i < total; i += st) Z[i] = b[i & nmask];
}

// bos_emb[h] = W_emb[h, 0] + b_emb[h]
__global__ void bos_kernel(const float* __restrict__ W_emb, const float* __restrict__ b_emb,
                           float* __restrict__ bos) {
  int h = threadIdx.x;
  if (h < NHID) bos[h] = W_emb[(long)h * NTOK] + b_emb[h];
}

// zbos[n] = sum_k W2ih0[n, k] * bos[k]   (k < 512, row stride 1024)
__global__ void zbos_kernel(const float* __restrict__ W2ih0, const float* __restrict__ bos,
                            float* __restrict__ zbos) {
  int n = blockIdx.x * blockDim.x + threadIdx.x;
  if (n >= H4) return;
  const float* row = W2ih0 + (long)n * (2 * NHID);
  float s = 0.0f;
  for (int k = 0; k < NHID; k++) s += row[k] * bos[k];
  zbos[n] = s;
}

// ---------------------------------------------------------------------------
// Generic bf16 WMMA GEMM:  C[M,N] = A[M,K] @ B[N, bcol:bcol+K]^T + bias[n]
//                          (+ extra[n] for rows >= splitRow)
// Per-wave tile: 16 (M) x 64 (N), K-loop in steps of 32, branch-free:
// B must be padded to a multiple of 64 rows (caller guarantees), so the
// K-loop has no guards; only the fp32 stores are masked by n < N.
// Work distribution is scalarized via readfirstlane (SALU control flow only).
// ---------------------------------------------------------------------------
__global__ void __launch_bounds__(256)
gemm_bf16_wmma(const __bf16* __restrict__ A, int K,
               const __bf16* __restrict__ B, int ldb, int bcol,
               const float* __restrict__ bias,
               const float* __restrict__ extra, int splitRow,
               float* __restrict__ C, int ldc, int M, int N) {
  const int lane  = threadIdx.x & 31;
  // wave index is uniform within the wave: make it scalar for SALU control flow
  const int wid   = __builtin_amdgcn_readfirstlane(
                      (int)((blockIdx.x * blockDim.x + threadIdx.x) >> 5));
  const int nWav  = (gridDim.x * blockDim.x) >> 5;
  const int Mt    = M >> 4;
  const int Nt    = (N + 15) >> 4;
  const int NG    = (Nt + 3) >> 2;       // 64-column groups (all full, B padded)
  const int total = Mt * NG;
  const int rowA  = lane & 15;           // A row within M-tile
  const int kselA = (lane >> 4) * 8;     // A K-subselect
  const int colB  = lane & 15;           // B col within N-tile
  const int kselB = (lane >> 4) * 8;     // B K-subselect

  for (int w = wid; w < total; w += nWav) {
    const int mt = w / NG;               // scalar
    const int ng = w % NG;               // scalar

    f32x8 acc0, acc1, acc2, acc3;
#pragma unroll
    for (int e = 0; e < 8; e++) { acc0[e] = 0.0f; acc1[e] = 0.0f; acc2[e] = 0.0f; acc3[e] = 0.0f; }

    const __bf16* Arow  = A + (long)(mt * 16 + rowA) * K + kselA;
    const __bf16* Brow0 = B + (long)(ng * 64 + colB) * ldb + bcol + kselB;
    const long bstep = (long)16 * ldb;   // 16 rows per ntile

#pragma unroll 2
    for (int kb = 0; kb < K; kb += 32) {
      bf16x8 a0  = *(const bf16x8*)(Arow + kb);
      bf16x8 a1  = *(const bf16x8*)(Arow + kb + 16);
      bf16x8 b00 = *(const bf16x8*)(Brow0 + kb);
      bf16x8 b01 = *(const bf16x8*)(Brow0 + kb + 16);
      bf16x8 b10 = *(const bf16x8*)(Brow0 + bstep + kb);
      bf16x8 b11 = *(const bf16x8*)(Brow0 + bstep + kb + 16);
      bf16x8 b20 = *(const bf16x8*)(Brow0 + 2 * bstep + kb);
      bf16x8 b21 = *(const bf16x8*)(Brow0 + 2 * bstep + kb + 16);
      bf16x8 b30 = *(const bf16x8*)(Brow0 + 3 * bstep + kb);
      bf16x8 b31 = *(const bf16x8*)(Brow0 + 3 * bstep + kb + 16);
      bf16x16 af = join16(a0, a1);
      acc0 = __builtin_amdgcn_wmma_f32_16x16x32_bf16(
          false, af, false, join16(b00, b01), (short)0, acc0, false, false);
      acc1 = __builtin_amdgcn_wmma_f32_16x16x32_bf16(
          false, af, false, join16(b10, b11), (short)0, acc1, false, false);
      acc2 = __builtin_amdgcn_wmma_f32_16x16x32_bf16(
          false, af, false, join16(b20, b21), (short)0, acc2, false, false);
      acc3 = __builtin_amdgcn_wmma_f32_16x16x32_bf16(
          false, af, false, join16(b30, b31), (short)0, acc3, false, false);
    }

    const int mbase = mt * 16 + (lane >> 4) * 8;
    const int nbase = ng * 64 + (lane & 15);
    const f32x8 accs[4] = {acc0, acc1, acc2, acc3};
#pragma unroll
    for (int q = 0; q < 4; q++) {
      const int n = nbase + q * 16;
      if (n >= N) continue;              // divergent only around stores
      const float bn = bias ? bias[n] : 0.0f;
      const float ex = extra ? extra[n] : 0.0f;
#pragma unroll
      for (int e = 0; e < 8; e++) {
        const int m = mbase + e;
        float v = accs[q][e] + bn + ((extra && m >= splitRow) ? ex : 0.0f);
        C[(long)m * ldc + n] = v;
      }
    }
  }
}

// ---------------------------------------------------------------------------
// Persistent single-workgroup LSTM recurrence over T steps.
//   z[t] = Zx[t] + h @ Whh^T ; gates i,f,g,o ; c,h update; Y[t] = h (bf16)
// 512 threads = 16 waves; wave owns 4 hidden tiles (1 M-tile x 4 J-tiles),
// each hidden tile needs 4 gate GEMM tiles. h lives in LDS (bf16, A-matrix);
// c lives in VGPRs (ownership is wave-stable across steps).
// ---------------------------------------------------------------------------
__global__ void __launch_bounds__(512)
lstm_seq(const float* __restrict__ Zx,      // [T][32][2048]
         const __bf16* __restrict__ Whh,    // [2048][512] bf16
         const float* __restrict__ h0,      // [32][512]
         const float* __restrict__ c0,      // [32][512]
         __bf16* __restrict__ Y,            // [T][32][512]
         int T) {
  __shared__ __align__(16) __bf16 hA[BSZ * NHID];   // 32 KB

  const int tid    = threadIdx.x;
  const int lane   = tid & 31;
  const int wave   = __builtin_amdgcn_readfirstlane(tid >> 5);  // 0..15, scalar
  const int mtile  = wave >> 3;       // 0..1  (M rows 0..15 / 16..31)
  const int jgroup = wave & 7;        // 0..7
  const int nsel   = lane & 15;
  const int rsel   = (lane >> 4) * 8;
  const int kselA  = (lane >> 4) * 8;
  const int kselB  = (lane >> 4) * 8;

  // init h into LDS (bf16)
  for (int i = tid; i < BSZ * NHID; i += 512) hA[i] = f2bf(h0[i]);

  // init c into registers (same element mapping as C-fragment)
  float creg[4][8];
#pragma unroll
  for (int q = 0; q < 4; q++) {
    const int j = (jgroup * 4 + q) * 16 + nsel;
#pragma unroll
    for (int e = 0; e < 8; e++) {
      const int m = mtile * 16 + rsel + e;
      creg[q][e] = c0[m * NHID + j];
    }
  }
  __syncthreads();

  for (int t = 0; t < T; t++) {
    const float* Zt = Zx + (long)t * BSZ * H4;
    float hn[4][8];

#pragma unroll
    for (int q = 0; q < 4; q++) {
      const int j = (jgroup * 4 + q) * 16 + nsel;

      f32x8 z[4];  // i, f, g, o accumulators, seeded with Zx
#pragma unroll
      for (int e = 0; e < 8; e++) {
        const int m = mtile * 16 + rsel + e;
        const float* zp = Zt + (long)m * H4 + j;
        z[0][e] = zp[0];
        z[1][e] = zp[NHID];
        z[2][e] = zp[2 * NHID];
        z[3][e] = zp[3 * NHID];
      }

      const __bf16* hrow = hA + (mtile * 16 + (lane & 15)) * NHID + kselA;
      const __bf16* brow = Whh + (long)j * NHID + kselB;
#pragma unroll 2
      for (int kb = 0; kb < NHID; kb += 32) {
        bf16x16 af = join16(*(const bf16x8*)(hrow + kb),
                            *(const bf16x8*)(hrow + kb + 16));
#pragma unroll
        for (int g = 0; g < 4; g++) {
          const __bf16* bp = brow + (long)g * NHID * NHID + kb;
          bf16x16 bfr = join16(*(const bf16x8*)bp, *(const bf16x8*)(bp + 16));
          z[g] = __builtin_amdgcn_wmma_f32_16x16x32_bf16(
              false, af, false, bfr, (short)0, z[g], false, false);
        }
      }

#pragma unroll
      for (int e = 0; e < 8; e++) {
        float cc = creg[q][e];
        const float iv = sigm(z[0][e]);
        const float fv = sigm(z[1][e]);
        const float gv = tanhf(z[2][e]);
        const float ov = sigm(z[3][e]);
        cc = fv * cc + iv * gv;
        creg[q][e] = cc;
        hn[q][e] = ov * tanhf(cc);
      }
      // write output sequence (bf16, feeds the next batched GEMM)
#pragma unroll
      for (int e = 0; e < 8; e++) {
        const int m = mtile * 16 + rsel + e;
        Y[(long)t * BSZ * NHID + m * NHID + j] = f2bf(hn[q][e]);
      }
    }

    __syncthreads();  // everyone done reading old h
#pragma unroll
    for (int q = 0; q < 4; q++) {
      const int j = (jgroup * 4 + q) * 16 + nsel;
#pragma unroll
      for (int e = 0; e < 8; e++) {
        const int m = mtile * 16 + rsel + e;
        hA[m * NHID + j] = f2bf(hn[q][e]);
      }
    }
    __syncthreads();  // new h visible to all
  }
}

// ---------------------------------------------------------------------------
// log-softmax over V per row (two kernels, in place on d_out)
// ---------------------------------------------------------------------------
__global__ void logsm_stats(const float* __restrict__ L, float* __restrict__ stats, int V) {
  const int r = blockIdx.x;
  const float* row = L + (long)r * V;
  float m = -__builtin_inff(), s = 0.0f;
  for (int v = threadIdx.x; v < V; v += blockDim.x) {
    const float x = row[v];
    const float nm = fmaxf(m, x);
    s = s * __expf(m - nm) + __expf(x - nm);
    m = nm;
  }
  __shared__ float sm[256], ss[256];
  sm[threadIdx.x] = m;
  ss[threadIdx.x] = s;
  __syncthreads();
  for (int o = 128; o > 0; o >>= 1) {
    if ((int)threadIdx.x < o) {
      const float m1 = sm[threadIdx.x], s1 = ss[threadIdx.x];
      const float m2 = sm[threadIdx.x + o], s2 = ss[threadIdx.x + o];
      const float nm = fmaxf(m1, m2);
      sm[threadIdx.x] = nm;
      ss[threadIdx.x] = s1 * __expf(m1 - nm) + s2 * __expf(m2 - nm);
    }
    __syncthreads();
  }
  if (threadIdx.x == 0) {
    stats[2 * r] = sm[0];
    stats[2 * r + 1] = __logf(ss[0]);
  }
}

__global__ void logsm_apply(float* __restrict__ L, const float* __restrict__ stats, int V) {
  const int r = blockIdx.y;
  const float m = stats[2 * r];
  const float ls = stats[2 * r + 1];
  float* row = L + (long)r * V;
  for (int v = blockIdx.x * blockDim.x + threadIdx.x; v < V; v += gridDim.x * blockDim.x)
    row[v] = row[v] - m - ls;
}

// ---------------------------------------------------------------------------
// host side
// ---------------------------------------------------------------------------
extern "C" void kernel_launch(void* const* d_in, const int* in_sizes, int n_in,
                              void* d_out, int out_size, void* d_ws, size_t ws_size,
                              hipStream_t stream) {
  (void)in_sizes; (void)n_in; (void)out_size; (void)ws_size;
  const float* inp     = (const float*)d_in[0];
  const float* hid1_h  = (const float*)d_in[1];
  const float* hid1_c  = (const float*)d_in[2];
  const float* hid2_h  = (const float*)d_in[3];
  const float* hid2_c  = (const float*)d_in[4];
  const float* l1_Wih0 = (const float*)d_in[5];
  const float* l1_Whh0 = (const float*)d_in[6];
  const float* l1_b0   = (const float*)d_in[7];
  const float* l1_Wih1 = (const float*)d_in[8];
  const float* l1_Whh1 = (const float*)d_in[9];
  const float* l1_b1   = (const float*)d_in[10];
  const float* l2_Wih0 = (const float*)d_in[11];
  const float* l2_Whh0 = (const float*)d_in[12];
  const float* l2_b0   = (const float*)d_in[13];
  const float* l2_Wih1 = (const float*)d_in[14];
  const float* l2_Whh1 = (const float*)d_in[15];
  const float* l2_b1   = (const float*)d_in[16];
  const float* W_out   = (const float*)d_in[17];
  const float* b_out   = (const float*)d_in[18];
  const float* W_emb   = (const float*)d_in[19];
  const float* b_emb   = (const float*)d_in[20];
  float* out = (float*)d_out;

  char* wsp = (char*)d_ws;
  auto alloc = [&](size_t bytes) -> char* {
    char* p = wsp;
    wsp += (bytes + 255) & ~(size_t)255;
    return p;
  };

  __bf16* inp_bf   = (__bf16*)alloc((size_t)T_ENC * BSZ * NINP * 2);
  __bf16* W1ih0_bf = (__bf16*)alloc((size_t)H4 * NINP * 2);
  __bf16* W1hh0_bf = (__bf16*)alloc((size_t)H4 * NHID * 2);
  __bf16* W1ih1_bf = (__bf16*)alloc((size_t)H4 * NHID * 2);
  __bf16* W1hh1_bf = (__bf16*)alloc((size_t)H4 * NHID * 2);
  __bf16* W2ih0_bf = (__bf16*)alloc((size_t)H4 * 2 * NHID * 2);
  __bf16* W2hh0_bf = (__bf16*)alloc((size_t)H4 * NHID * 2);
  __bf16* W2ih1_bf = (__bf16*)alloc((size_t)H4 * NHID * 2);
  __bf16* W2hh1_bf = (__bf16*)alloc((size_t)H4 * NHID * 2);
  __bf16* Wout_bf  = (__bf16*)alloc((size_t)NTOK_PAD * NHID * 2);
  float*  Z1_0     = (float*)alloc((size_t)T_ALL * BSZ * H4 * 4);
  float*  Z1_1     = (float*)alloc((size_t)T_ALL * BSZ * H4 * 4);
  float*  Z2_0     = (float*)alloc((size_t)T_ALL * BSZ * H4 * 4);
  float*  Z2_1     = (float*)alloc((size_t)T_ALL * BSZ * H4 * 4);
  __bf16* Y1_0     = (__bf16*)alloc((size_t)T_ALL * BSZ * NHID * 2);
  __bf16* Y1_1     = (__bf16*)alloc((size_t)T_ALL * BSZ * NHID * 2);
  __bf16* Y2_0     = (__bf16*)alloc((size_t)T_ALL * BSZ * NHID * 2);
  __bf16* Y2_1     = (__bf16*)alloc((size_t)T_ALL * BSZ * NHID * 2);
  float*  bos      = (float*)alloc((size_t)NHID * 4);
  float*  zbos     = (float*)alloc((size_t)H4 * 4);
  float*  stats    = (float*)alloc((size_t)T_DEC * BSZ * 2 * 4);

  // --- precision conversion ---
  cvt_bf16<<<2048, 256, 0, stream>>>(inp, inp_bf, (long)T_ENC * BSZ * NINP);
  cvt_bf16<<<1024, 256, 0, stream>>>(l1_Wih0, W1ih0_bf, (long)H4 * NINP);
  cvt_bf16<<<256, 256, 0, stream>>>(l1_Whh0, W1hh0_bf, (long)H4 * NHID);
  cvt_bf16<<<256, 256, 0, stream>>>(l1_Wih1, W1ih1_bf, (long)H4 * NHID);
  cvt_bf16<<<256, 256, 0, stream>>>(l1_Whh1, W1hh1_bf, (long)H4 * NHID);
  cvt_bf16<<<512, 256, 0, stream>>>(l2_Wih0, W2ih0_bf, (long)H4 * 2 * NHID);
  cvt_bf16<<<256, 256, 0, stream>>>(l2_Whh0, W2hh0_bf, (long)H4 * NHID);
  cvt_bf16<<<256, 256, 0, stream>>>(l2_Wih1, W2ih1_bf, (long)H4 * NHID);
  cvt_bf16<<<256, 256, 0, stream>>>(l2_Whh1, W2hh1_bf, (long)H4 * NHID);
  cvt_bf16_pad_rows<<<2048, 256, 0, stream>>>(W_out, Wout_bf, NTOK, NHID, NTOK_PAD);

  // --- constant decode-side input contributions ---
  bos_kernel<<<1, 512, 0, stream>>>(W_emb, b_emb, bos);
  zbos_kernel<<<8, 256, 0, stream>>>(l2_Wih0, bos, zbos);

  // --- lstm1 layer0: batched input GEMM (encode), bias-only (decode), recurrence ---
  gemm_bf16_wmma<<<1024, 256, 0, stream>>>(inp_bf, NINP, W1ih0_bf, NINP, 0,
                                           l1_b0, nullptr, 0,
                                           Z1_0, H4, T_ENC * BSZ, H4);
  fill_bias<<<1024, 256, 0, stream>>>(Z1_0 + (long)T_ENC * BSZ * H4, l1_b0,
                                      (long)T_DEC * BSZ * H4, H4 - 1);
  lstm_seq<<<1, 512, 0, stream>>>(Z1_0, W1hh0_bf, hid1_h, hid1_c, Y1_0, T_ALL);

  // --- lstm1 layer1 ---
  gemm_bf16_wmma<<<1024, 256, 0, stream>>>(Y1_0, NHID, W1ih1_bf, NHID, 0,
                                           l1_b1, nullptr, 0,
                                           Z1_1, H4, T_ALL * BSZ, H4);
  lstm_seq<<<1, 512, 0, stream>>>(Z1_1, W1hh1_bf, hid1_h + BSZ * NHID,
                                  hid1_c + BSZ * NHID, Y1_1, T_ALL);

  // --- lstm2 layer0: second half of concat input is Y1_1; first half is 0 (enc) / bos (dec) ---
  gemm_bf16_wmma<<<1024, 256, 0, stream>>>(Y1_1, NHID, W2ih0_bf, 2 * NHID, NHID,
                                           l2_b0, zbos, T_ENC * BSZ,
                                           Z2_0, H4, T_ALL * BSZ, H4);
  lstm_seq<<<1, 512, 0, stream>>>(Z2_0, W2hh0_bf, hid2_h, hid2_c, Y2_0, T_ALL);

  // --- lstm2 layer1 ---
  gemm_bf16_wmma<<<1024, 256, 0, stream>>>(Y2_0, NHID, W2ih1_bf, NHID, 0,
                                           l2_b1, nullptr, 0,
                                           Z2_1, H4, T_ALL * BSZ, H4);
  lstm_seq<<<1, 512, 0, stream>>>(Z2_1, W2hh1_bf, hid2_h + BSZ * NHID,
                                  hid2_c + BSZ * NHID, Y2_1, T_ALL);

  // --- output projection over vocab: decode part of Y2_1 only ---
  gemm_bf16_wmma<<<2048, 256, 0, stream>>>(Y2_1 + (long)T_ENC * BSZ * NHID, NHID,
                                           Wout_bf, NHID, 0,
                                           b_out, nullptr, 0,
                                           out, NTOK, T_DEC * BSZ, NTOK);

  // --- log-softmax, in place on d_out ---
  logsm_stats<<<T_DEC * BSZ, 256, 0, stream>>>(out, stats, NTOK);
  logsm_apply<<<dim3(64, T_DEC * BSZ), 256, 0, stream>>>(out, stats, NTOK);
}
